// MultiHeadAttention_283467842584
// MI455X (gfx1250) — matrix operand order repaired and checked by reading, not compiled, not run on hardware
//
#include <hip/hip_runtime.h>
#include <cstdint>
#include <cstddef>

// ---------------------------------------------------------------------------
// Types for WMMA
// ---------------------------------------------------------------------------
typedef __bf16  bf16_t;
typedef __attribute__((ext_vector_type(16))) __bf16        v16bf;
typedef __attribute__((ext_vector_type(8)))  float         v8f;
typedef __attribute__((ext_vector_type(4)))  unsigned int  u32x4;
typedef __attribute__((ext_vector_type(8)))  unsigned int  u32x8;

#define SCALE_F 0.125f   // 64^-0.5
#define NSEQ    2048
#define NHEAD   12
#define HDIM    64
#define DMODEL  768
#define QKVW    2304     // 3*H*HD
#define BATCH   4
#define MROWS   (BATCH * NSEQ)   // 8192
#define BSTRIDE 40       // LDS B-stage row stride (ushorts): 80B, 16B-aligned, conflict-free

// ---------------------------------------------------------------------------
// fp32 -> bf16 (round to nearest even), stored as raw u16
// ---------------------------------------------------------------------------
__device__ __forceinline__ unsigned short f2bf(float f) {
    unsigned int u = __builtin_bit_cast(unsigned int, f);
    unsigned int r = u + 0x7FFFu + ((u >> 16) & 1u);
    return (unsigned short)(r >> 16);
}

// ---------------------------------------------------------------------------
// CDNA5 async global->LDS copy (ASYNCcnt-tracked, §15.18.3 opcode 98)
// 16 bytes per lane. LDS offset = low 32 bits of generic shared pointer.
// ---------------------------------------------------------------------------
__device__ __forceinline__ void async_copy_b128(unsigned short* lds_dst,
                                                const unsigned short* gsrc) {
    unsigned int ldsa = (unsigned int)(uintptr_t)lds_dst;
    asm volatile("global_load_async_to_lds_b128 %0, %1, off"
                 :: "v"(ldsa), "v"(gsrc) : "memory");
}
__device__ __forceinline__ void wait_async0() {
    asm volatile("s_wait_asynccnt 0x0" ::: "memory");
}

// ---------------------------------------------------------------------------
// Fragment loaders (layouts per CDNA5 ISA 7.12.2, wave32)
// A 16x32 bf16: lane(0..15)=row M, halves split K: elems[0..7]=K[kb..kb+8),
//               elems[8..15]=K[kb+16..kb+24), kb = k0 + 8*(lane>>4)
// B 32x16 bf16: lane%16 = col N, elems[0..7]=K[kb..kb+8), elems[8..15]=K[kb+16..kb+24), kb = k0+8*(lane>>4)
// ---------------------------------------------------------------------------
__device__ __forceinline__ v16bf load_a_frag(const unsigned short* base,
                                             int stride, int row, int kb) {
    const unsigned short* p = base + (size_t)row * stride + kb;
    u32x4 lo = *(const u32x4*)(p);
    u32x4 hi = *(const u32x4*)(p + 16);
    u32x8 r;
    r[0] = lo[0]; r[1] = lo[1]; r[2] = lo[2]; r[3] = lo[3];
    r[4] = hi[0]; r[5] = hi[1]; r[6] = hi[2]; r[7] = hi[3];
    return __builtin_bit_cast(v16bf, r);
}

__device__ __forceinline__ v16bf load_b_frag(const unsigned short* base,
                                             int stride, int col, int kb) {
    const unsigned short* p = base + (size_t)col * stride + kb;
    u32x4 lo = *(const u32x4*)(p);
    u32x4 hi = *(const u32x4*)(p + 16);
    u32x8 r;
    r[0] = lo[0]; r[1] = lo[1]; r[2] = lo[2]; r[3] = lo[3];
    r[4] = hi[0]; r[5] = hi[1]; r[6] = hi[2]; r[7] = hi[3];
    return __builtin_bit_cast(v16bf, r);
}

#define WMMA_BF16(a, b, c) \
    __builtin_amdgcn_wmma_f32_16x16x32_bf16(false, (a), false, (b), (short)0, (c), false, false)

// ---------------------------------------------------------------------------
// Elementwise convert fp32 -> bf16
// ---------------------------------------------------------------------------
__global__ void convert_bf16_kernel(const float* __restrict__ src,
                                    unsigned short* __restrict__ dst, int n) {
    int i = blockIdx.x * blockDim.x + threadIdx.x;
    if (i < n) dst[i] = f2bf(src[i]);
}

// src [rows][cols] fp32  ->  dst [cols][rows] bf16 (transposed)
__global__ void transpose_bf16_kernel(const float* __restrict__ src,
                                      unsigned short* __restrict__ dst,
                                      int rows, int cols) {
    int i = blockIdx.x * blockDim.x + threadIdx.x;
    if (i >= rows * cols) return;
    int r = i / cols, c = i % cols;
    dst[(size_t)c * rows + r] = f2bf(src[i]);
}

// ---------------------------------------------------------------------------
// WMMA GEMM:  out[M][Nout] = A[M][K](bf16) * BT[Nout][K](bf16)^T + bias
// block = 256 threads = 8 waves; wave tile 32x64; block tile 256x64.
// B tile (64 cols x 32 k) double-buffered in LDS via async global->LDS copies,
// prefetched one k-step ahead (ASYNCcnt + barrier sync).
// mode 0: write bf16 to outB;  mode 1: write fp32 to outF
// ---------------------------------------------------------------------------
__global__ __launch_bounds__(256, 1) void gemm_wmma_kernel(
    const unsigned short* __restrict__ A, int lda,
    const unsigned short* __restrict__ BT, int ldb,
    const float* __restrict__ bias,
    int K, int Nout,
    unsigned short* __restrict__ outB, float* __restrict__ outF, int mode) {

    __shared__ unsigned short bstage[2][64 * BSTRIDE];

    const int lane = threadIdx.x & 31;
    const int wave = threadIdx.x >> 5;
    const int half = lane >> 4;
    const int l15  = lane & 15;
    const int m0 = blockIdx.x * 256 + wave * 32;
    const int n0 = blockIdx.y * 64;

    // async-stage thread mapping: 256 threads x 16B cover the 64x32 B tile
    const int scol = threadIdx.x >> 2;        // 0..63  (column of B tile)
    const int sch  = (threadIdx.x & 3) * 8;   // 0,8,16,24 (k chunk of 8 ushorts)
    const unsigned short* bsrc = BT + (size_t)(n0 + scol) * ldb + sch;

    v8f acc[2][4];
#pragma unroll
    for (int mt = 0; mt < 2; ++mt)
#pragma unroll
        for (int nt = 0; nt < 4; ++nt) acc[mt][nt] = (v8f){};

    // prefetch k-step 0
    async_copy_b128(&bstage[0][scol * BSTRIDE + sch], bsrc);

    int buf = 0;
    for (int k0 = 0; k0 < K; k0 += 32) {
        wait_async0();        // own async copies into bstage[buf] landed
        __syncthreads();      // all waves' copies landed; prev reads of buf^1 done

        if (k0 + 32 < K)      // prefetch next k-step into the other buffer
            async_copy_b128(&bstage[buf ^ 1][scol * BSTRIDE + sch],
                            bsrc + (k0 + 32));

        v16bf a0 = load_a_frag(A, lda, m0 + l15,      k0 + 8 * half);
        v16bf a1 = load_a_frag(A, lda, m0 + 16 + l15, k0 + 8 * half);
#pragma unroll
        for (int nt = 0; nt < 4; ++nt) {
            v16bf b = load_b_frag(bstage[buf], BSTRIDE, nt * 16 + l15, 8 * half);
            acc[0][nt] = WMMA_BF16(a0, b, acc[0][nt]);
            acc[1][nt] = WMMA_BF16(a1, b, acc[1][nt]);
        }
        buf ^= 1;
    }

#pragma unroll
    for (int mt = 0; mt < 2; ++mt) {
#pragma unroll
        for (int nt = 0; nt < 4; ++nt) {
#pragma unroll
            for (int r = 0; r < 8; ++r) {
                int row = m0 + mt * 16 + r + 8 * half;
                int col = n0 + nt * 16 + l15;
                float v = acc[mt][nt][r] + bias[col];
                if (mode == 0) outB[(size_t)row * Nout + col] = f2bf(v);
                else           outF[(size_t)row * Nout + col] = v;
            }
        }
    }
}

// ---------------------------------------------------------------------------
// Flash attention core.
// Reference: scores = k @ q^T ; softmax over q-index ; * SCALE ; @ v
// => out[n] = SCALE * sum_m softmax_m(k[n].q[m]) v[m]   (K acts as "query")
// qkvb layout: [B*N][2304] bf16, columns = which*768 + h*64 + d
// ob   layout: [B*N][768]  bf16 (head-concatenated, ready for proj GEMM)
// grid(32, 12, 4), block 128 (4 waves); each wave owns 16 output (k) rows.
// ---------------------------------------------------------------------------
__global__ __launch_bounds__(128, 1) void attn_kernel(
    const unsigned short* __restrict__ qkvb,
    unsigned short* __restrict__ ob) {

    __shared__ unsigned short vt[64 * 72];      // V^T tile: [d][m], stride 72 (16B-aligned rows)
    __shared__ unsigned short pp[4 * 16 * 64];  // per-wave P tile, row-major [16][64]

    const int lane = threadIdx.x & 31;
    const int wave = threadIdx.x >> 5;
    const int half = lane >> 4;
    const int l15  = lane & 15;

    const int rt = blockIdx.x;   // k-row tile (64 rows)
    const int h  = blockIdx.y;
    const int b  = blockIdx.z;
    const int r0 = rt * 64 + wave * 16;

    const unsigned short* qbase = qkvb + (size_t)b * NSEQ * QKVW + h * HDIM;
    const unsigned short* kbase = qbase + DMODEL;       // which=1
    const unsigned short* vbase = qbase + 2 * DMODEL;   // which=2

    // K rows of this wave as A fragments, resident for whole loop (HD=64 -> 2 k-steps)
    v16bf ak[2];
#pragma unroll
    for (int ks = 0; ks < 2; ++ks)
        ak[ks] = load_a_frag(kbase, QKVW, r0 + l15, ks * 32 + 8 * half);

    v8f o[4];
#pragma unroll
    for (int t = 0; t < 4; ++t) o[t] = (v8f){};
    float mst[8], lst[8];
#pragma unroll
    for (int r = 0; r < 8; ++r) { mst[r] = -1e30f; lst[r] = 0.0f; }

    for (int j = 0; j < NSEQ / 64; ++j) {
        const int c0 = j * 64;

        // ---- stage V^T tile into LDS (coalesced 16B global loads, 2B ds scatter)
#pragma unroll
        for (int q = 0; q < 4; ++q) {
            int c  = q * 128 + threadIdx.x;   // chunk 0..511 of 8 ushorts
            int m  = c >> 3;                  // V row within tile
            int d0 = (c & 7) * 8;             // V col start
            u32x4 val = *(const u32x4*)(vbase + (size_t)(c0 + m) * QKVW + d0);
            union { u32x4 v; unsigned short u[8]; } tv; tv.v = val;
#pragma unroll
            for (int e = 0; e < 8; ++e) vt[(d0 + e) * 72 + m] = tv.u[e];
        }
        __syncthreads();

        // ---- S = K_tile (16xHD) @ Q_tile^T  : B-operand columns are Q rows (contiguous)
        v8f s[4];
#pragma unroll
        for (int t = 0; t < 4; ++t) s[t] = (v8f){};
#pragma unroll
        for (int nt = 0; nt < 4; ++nt) {
#pragma unroll
            for (int ks = 0; ks < 2; ++ks) {
                v16bf qb = load_b_frag(qbase, QKVW, c0 + nt * 16 + l15, ks * 32 + 8 * half);
                s[nt] = WMMA_BF16(ak[ks], qb, s[nt]);
            }
        }

        // ---- online softmax (row-wise: reduce across 16 lanes of this half)
#pragma unroll
        for (int r = 0; r < 8; ++r) {
            float cur = fmaxf(fmaxf(s[0][r], s[1][r]), fmaxf(s[2][r], s[3][r]));
            cur = fmaxf(cur, __shfl_xor(cur, 1, 32));
            cur = fmaxf(cur, __shfl_xor(cur, 2, 32));
            cur = fmaxf(cur, __shfl_xor(cur, 4, 32));
            cur = fmaxf(cur, __shfl_xor(cur, 8, 32));
            float nm   = fmaxf(mst[r], cur);
            float corr = __expf(mst[r] - nm);
            mst[r] = nm;
            float rs = 0.0f;
#pragma unroll
            for (int nt = 0; nt < 4; ++nt) {
                float p = __expf(s[nt][r] - nm);
                s[nt][r] = p;
                rs += p;
            }
            rs += __shfl_xor(rs, 1, 32);
            rs += __shfl_xor(rs, 2, 32);
            rs += __shfl_xor(rs, 4, 32);
            rs += __shfl_xor(rs, 8, 32);
            lst[r] = lst[r] * corr + rs;
#pragma unroll
            for (int nt = 0; nt < 4; ++nt) o[nt][r] *= corr;
        }

        // ---- P (C-layout f32) -> bf16 A-layout via per-wave LDS round trip
        unsigned short* pw = pp + wave * 16 * 64;
#pragma unroll
        for (int nt = 0; nt < 4; ++nt)
#pragma unroll
            for (int r = 0; r < 8; ++r)
                pw[(r + 8 * half) * 64 + nt * 16 + l15] = f2bf(s[nt][r]);

        v16bf pa[2];
#pragma unroll
        for (int ks = 0; ks < 2; ++ks)
            pa[ks] = load_a_frag(pw, 64, l15, ks * 32 + 8 * half);

        // ---- O += P (16x64) @ V (64xHD), V^T rows from LDS
#pragma unroll
        for (int nt = 0; nt < 4; ++nt) {
#pragma unroll
            for (int ks = 0; ks < 2; ++ks) {
                v16bf vb = load_b_frag(vt, 72, nt * 16 + l15, ks * 32 + 8 * half);
                o[nt] = WMMA_BF16(pa[ks], vb, o[nt]);
            }
        }
        __syncthreads();
    }

    // ---- epilogue: normalize, apply post-softmax SCALE, store head-concat bf16
#pragma unroll
    for (int nt = 0; nt < 4; ++nt) {
#pragma unroll
        for (int r = 0; r < 8; ++r) {
            int row = r0 + r + 8 * half;
            int d   = nt * 16 + l15;
            float v = o[nt][r] / lst[r] * SCALE_F;
            ob[((size_t)b * NSEQ + row) * DMODEL + h * HDIM + d] = f2bf(v);
        }
    }
}

// ---------------------------------------------------------------------------
// Host launcher
// ---------------------------------------------------------------------------
extern "C" void kernel_launch(void* const* d_in, const int* in_sizes, int n_in,
                              void* d_out, int out_size, void* d_ws, size_t ws_size,
                              hipStream_t stream) {
    (void)in_sizes; (void)n_in; (void)out_size; (void)ws_size;

    const float* x      = (const float*)d_in[0];   // [B,N,D]
    const float* W_qkv  = (const float*)d_in[1];   // [D, 2304]
    const float* b_qkv  = (const float*)d_in[2];   // [2304]
    const float* W_proj = (const float*)d_in[3];   // [768, 768]
    const float* b_proj = (const float*)d_in[4];   // [768]
    float* out = (float*)d_out;                    // [B,N,768]

    // workspace carve-up (bf16 stored as u16)
    size_t off = 0;
    auto carve = [&](size_t bytes) {
        void* p = (char*)d_ws + off;
        off += (bytes + 255) & ~(size_t)255;
        return p;
    };
    unsigned short* xb     = (unsigned short*)carve((size_t)MROWS * DMODEL * 2);
    unsigned short* wqkvT  = (unsigned short*)carve((size_t)QKVW  * DMODEL * 2);
    unsigned short* wprojT = (unsigned short*)carve((size_t)DMODEL * DMODEL * 2);
    unsigned short* qkvb   = (unsigned short*)carve((size_t)MROWS * QKVW * 2);
    unsigned short* ob     = (unsigned short*)carve((size_t)MROWS * DMODEL * 2);

    // 1) precision conversion (+ weight transposes for contiguous B-fragments)
    {
        int n = MROWS * DMODEL;
        convert_bf16_kernel<<<(n + 255) / 256, 256, 0, stream>>>(x, xb, n);
    }
    {
        int n = DMODEL * QKVW;
        transpose_bf16_kernel<<<(n + 255) / 256, 256, 0, stream>>>(W_qkv, wqkvT, DMODEL, QKVW);
    }
    {
        int n = DMODEL * DMODEL;
        transpose_bf16_kernel<<<(n + 255) / 256, 256, 0, stream>>>(W_proj, wprojT, DMODEL, DMODEL);
    }

    // 2) QKV projection: [8192,768] x [768,2304] + bias -> bf16 qkv
    {
        dim3 grid(MROWS / 256, QKVW / 64);
        gemm_wmma_kernel<<<grid, 256, 0, stream>>>(xb, DMODEL, wqkvT, DMODEL,
                                                   b_qkv, DMODEL, QKVW,
                                                   qkvb, nullptr, 0);
    }

    // 3) fused flash attention (K@Q^T, online softmax, @V, *SCALE)
    {
        dim3 grid(NSEQ / 64, NHEAD, BATCH);
        attn_kernel<<<grid, 128, 0, stream>>>(qkvb, ob);
    }

    // 4) output projection: [8192,768] x [768,768] + bias -> fp32 out
    {
        dim3 grid(MROWS / 256, DMODEL / 64);
        gemm_wmma_kernel<<<grid, 256, 0, stream>>>(ob, DMODEL, wprojT, DMODEL,
                                                   b_proj, DMODEL, DMODEL,
                                                   nullptr, out, 1);
    }
}
